// MultiHeadCrossAttention_24240795419422
// MI455X (gfx1250) — matrix operand order repaired and checked by reading, not compiled, not run on hardware
//
#include <hip/hip_runtime.h>

// ---------------------------------------------------------------------------
// MultiHeadCrossAttention for MI455X (gfx1250, wave32, WMMA bf16 16x16x32)
//
// All heavy math runs through v_wmma_f32_16x16x32_bf16 (f32 accumulate).
// Logits are computed TRANSPOSED so both softmaxes (over the S axis) become
// row-wise and fuse into the GEMM kernels via a 16x1024 f32 LDS tile, and so
// the dominant 96x(1024^3) einsum gets contiguous-K operands on both sides.
//
// Round 4: the dominant einsum (K3) now computes 32 output rows per WG so
// every B fragment feeds two WMMAs (16 WMMAs per 20 loads), halving L2
// traffic for the 2MB/bh B matrix. Ping-pong double buffering + sched-group
// pinning retained; __launch_bounds__(256,1) relaxes the occupancy target so
// the register allocator keeps the double buffer live without spilling.
// ---------------------------------------------------------------------------

typedef __attribute__((ext_vector_type(16))) __bf16       bf16x16;
typedef __attribute__((ext_vector_type(8)))  float        v8f;
typedef __attribute__((ext_vector_type(4)))  unsigned int u32x4;

#if defined(__has_builtin)
#if __has_builtin(__builtin_amdgcn_sched_group_barrier)
#define SCHED_GROUP(mask, size, id) __builtin_amdgcn_sched_group_barrier((mask), (size), (id))
#endif
#endif
#ifndef SCHED_GROUP
#define SCHED_GROUP(mask, size, id)
#endif
// masks: 0x008 = MFMA/WMMA, 0x020 = VMEM read, 0x200 = DS write

union FragU {
  u32x4   q[2];
  bf16x16 v;
};

__device__ __forceinline__ unsigned short f2bf(float f) {
  union { float f; unsigned int u; } x;
  x.f = f;
  unsigned int u = x.u;
  unsigned int r = u + 0x7FFFu + ((u >> 16) & 1u);  // round-to-nearest-even
  return (unsigned short)(r >> 16);
}

// A fragment: 16x32 bf16, rows of A, contiguous k. ISA layout:
// lane m=lane&15, half=lane>>4; VGPR0-3: k = half*8+0..7 ; VGPR4-7: k = 16+half*8+0..7
__device__ __forceinline__ bf16x16 load_frag_a(const unsigned short* __restrict__ base,
                                               int ld, int lane) {
  const int m = lane & 15, half = lane >> 4;
  const unsigned short* r = base + (size_t)m * ld + half * 8;
  FragU f;
  f.q[0] = *(const u32x4*)(r);        // k = half*8 .. half*8+7
  f.q[1] = *(const u32x4*)(r + 16);   // k = 16+half*8 .. +7
  return f.v;
}

// B fragment: 32x16 bf16 fed from rows of B^T (i.e. columns of B), contiguous k.
// lane n=lane&15, half=lane>>4; VGPR0-3: k = half*8+0..7 ; VGPR4-7: k = 16+half*8+0..7
__device__ __forceinline__ bf16x16 load_frag_b(const unsigned short* __restrict__ base,
                                               int ld, int lane) {
  const int n = lane & 15, half = lane >> 4;
  const unsigned short* r = base + (size_t)n * ld + half * 8;
  FragU f;
  f.q[0] = *(const u32x4*)(r);
  f.q[1] = *(const u32x4*)(r + 16);
  return f.v;
}

__device__ __forceinline__ v8f wmma_bf16(bf16x16 a, bf16x16 b, v8f c) {
  return __builtin_amdgcn_wmma_f32_16x16x32_bf16(false, a, false, b, (short)0, c,
                                                 false, false);
}

// ---------------------------------------------------------------------------
// K0: f32 -> bf16 conversion
// ---------------------------------------------------------------------------
__global__ void k_f32_to_bf16(const float* __restrict__ in,
                              unsigned short* __restrict__ out, int n) {
  int i = blockIdx.x * 256 + threadIdx.x;
  if (i < n) out[i] = f2bf(in[i]);
}

// ---------------------------------------------------------------------------
// K1/K5: C[m,n] = (sum_k A[m,k]*Bt[n,k] + bias[n]) * scale
// A: [M,K] bf16 row-major, Bt: [N,K] bf16 row-major (rows = columns of B).
// N must be 768: 8 waves x 96 cols. grid.x = M/16. 256 threads.
// Ping-pong double-buffered over k (K assumed a multiple of 64).
// ---------------------------------------------------------------------------
template <bool OUT_F32>
__global__ void __launch_bounds__(256)
k_gemm768(const unsigned short* __restrict__ A,
          const unsigned short* __restrict__ Bt,
          const float* __restrict__ bias, float scale,
          unsigned short* __restrict__ outBf, float* __restrict__ outF, int K) {
  const int lane = threadIdx.x & 31;
  const int wave = threadIdx.x >> 5;
  const int row0 = blockIdx.x * 16;
  const v8f zero = {0.f, 0.f, 0.f, 0.f, 0.f, 0.f, 0.f, 0.f};
  v8f acc[6];
#pragma unroll
  for (int c = 0; c < 6; ++c) acc[c] = zero;

  const unsigned short* Arow = A + (size_t)row0 * K;
  const unsigned short* Bw = Bt + (size_t)(wave * 96) * K;

  // prologue: buffer 0 <- k = 0
  bf16x16 a0 = load_frag_a(Arow, K, lane);
  bf16x16 b0[6], b1[6];
  bf16x16 a1;
#pragma unroll
  for (int c = 0; c < 6; ++c) b0[c] = load_frag_b(Bw + (size_t)c * 16 * K, K, lane);

  for (int k0 = 0; k0 < K; k0 += 64) {
    const int k1 = k0 + 32;
    // issue loads for k1 into buffer 1
    a1 = load_frag_a(Arow + k1, K, lane);
#pragma unroll
    for (int c = 0; c < 6; ++c)
      b1[c] = load_frag_b(Bw + (size_t)c * 16 * K + k1, K, lane);
    // compute with buffer 0 (k0)
#pragma unroll
    for (int c = 0; c < 6; ++c) acc[c] = wmma_bf16(a0, b0[c], acc[c]);
    // issue loads for k0+64 into buffer 0
    const int k2 = k0 + 64;
    if (k2 < K) {
      a0 = load_frag_a(Arow + k2, K, lane);
#pragma unroll
      for (int c = 0; c < 6; ++c)
        b0[c] = load_frag_b(Bw + (size_t)c * 16 * K + k2, K, lane);
    }
    // compute with buffer 1 (k1)
#pragma unroll
    for (int c = 0; c < 6; ++c) acc[c] = wmma_bf16(a1, b1[c], acc[c]);

    // pin the pipeline: loads(k1) | wmma(k0) | loads(k2) | wmma(k1)
    SCHED_GROUP(0x020, 14, 0);
    SCHED_GROUP(0x008, 6, 0);
    SCHED_GROUP(0x020, 14, 0);
    SCHED_GROUP(0x008, 6, 0);
  }

  const int n = lane & 15, half = lane >> 4;
#pragma unroll
  for (int c = 0; c < 6; ++c) {
    const int col = wave * 96 + c * 16 + n;
    const float bv = bias[col];
#pragma unroll
    for (int r = 0; r < 8; ++r) {
      const size_t row = (size_t)(row0 + half * 8 + r);
      const float v = (acc[c][r] + bv) * scale;
      if (OUT_F32) outF[row * 768 + col] = v;
      else         outBf[row * 768 + col] = f2bf(v);
    }
  }
}

// ---------------------------------------------------------------------------
// K2: fused transposed logits + softmax-over-S.
//   outT[bh][k][s] = softmax_s( sum_d Kmat[b][k][h*64+d] * Qmat[b][s][h*64+d] )
// grid = (64 k-blocks, 96 bh), 256 threads, 64KB LDS tile.
// A fragments are loop-invariant; B fragments ping-pong across column chunks.
// ---------------------------------------------------------------------------
__global__ void __launch_bounds__(256)
k_logits_softmax(const unsigned short* __restrict__ Kmat,
                 const unsigned short* __restrict__ Qmat,
                 unsigned short* __restrict__ outT) {
  __shared__ float tile[16][1024];
  const int lane = threadIdx.x & 31;
  const int wave = threadIdx.x >> 5;
  const int kb = blockIdx.x;
  const int bh = blockIdx.y;
  const int b = bh / 12, h = bh % 12;

  const unsigned short* Kb = Kmat + ((size_t)b * 1024 + kb * 16) * 768 + h * 64;
  const unsigned short* Qb = Qmat + (size_t)b * 1024 * 768 + h * 64;

  const bf16x16 a0 = load_frag_a(Kb + 0, 768, lane);   // d = 0..31
  const bf16x16 a1 = load_frag_a(Kb + 32, 768, lane);  // d = 32..63
  const v8f zero = {0.f, 0.f, 0.f, 0.f, 0.f, 0.f, 0.f, 0.f};
  const int n = lane & 15, half = lane >> 4;

  const int sBase = wave * 128;
  // prologue: chunk 0
  bf16x16 u0 = load_frag_b(Qb + (size_t)sBase * 768, 768, lane);
  bf16x16 u1 = load_frag_b(Qb + (size_t)sBase * 768 + 32, 768, lane);
  bf16x16 w0, w1;

#pragma unroll
  for (int c = 0; c < 8; c += 2) {
    const int sA = sBase + c * 16;
    const int sB = sA + 16;
    const int sC = sA + 32;
    // load chunk c+1
    w0 = load_frag_b(Qb + (size_t)sB * 768, 768, lane);
    w1 = load_frag_b(Qb + (size_t)sB * 768 + 32, 768, lane);
    // compute chunk c
    v8f acc = zero;
    acc = wmma_bf16(a0, u0, acc);
    acc = wmma_bf16(a1, u1, acc);
#pragma unroll
    for (int r = 0; r < 8; ++r) tile[half * 8 + r][sA + n] = acc[r];
    // load chunk c+2
    if (c + 2 < 8) {
      u0 = load_frag_b(Qb + (size_t)sC * 768, 768, lane);
      u1 = load_frag_b(Qb + (size_t)sC * 768 + 32, 768, lane);
    }
    // compute chunk c+1
    v8f acc2 = zero;
    acc2 = wmma_bf16(a0, w0, acc2);
    acc2 = wmma_bf16(a1, w1, acc2);
#pragma unroll
    for (int r = 0; r < 8; ++r) tile[half * 8 + r][sB + n] = acc2[r];

    // pin: loads(c+1) | wmma(c) | ds(c) | loads(c+2) | wmma(c+1) | ds(c+1)
    SCHED_GROUP(0x020, 4, 0);
    SCHED_GROUP(0x008, 2, 0);
    SCHED_GROUP(0x200, 8, 0);
    SCHED_GROUP(0x020, 4, 0);
    SCHED_GROUP(0x008, 2, 0);
    SCHED_GROUP(0x200, 8, 0);
  }
  __syncthreads();

  // row-wise softmax over 1024; 16 threads per row, shuffle reduce in half-wave
  const int row = threadIdx.x >> 4;
  const int sub = threadIdx.x & 15;
  float mx = -1e30f;
  for (int i = sub; i < 1024; i += 16) mx = fmaxf(mx, tile[row][i]);
  for (int m = 8; m >= 1; m >>= 1) mx = fmaxf(mx, __shfl_xor(mx, m, 32));
  float sum = 0.f;
  for (int i = sub; i < 1024; i += 16) {
    const float e = __expf(tile[row][i] - mx);
    tile[row][i] = e;
    sum += e;
  }
  for (int m = 8; m >= 1; m >>= 1) sum += __shfl_xor(sum, m, 32);
  const float inv = 1.0f / sum;
  unsigned short* orow = outT + ((size_t)bh * 1024 + kb * 16 + row) * 1024;
  for (int i = sub; i < 1024; i += 16) orow[i] = f2bf(tile[row][i] * inv);
}

// ---------------------------------------------------------------------------
// K3: SP2P[p,k] = softmax_k( sum_s L2PT[p,s] * L2SPT[k,s] )   (per bh)
// The dominant 96 x (1024^3) batched GEMM, fused with the row softmax.
// grid = (32 p-blocks of 32 rows, 96 bh), 256 threads, 64KB LDS tile.
// Two A-row blocks share every B fragment (16 WMMAs per 20 loads), halving
// L2 traffic vs 16-row blocking. Ping-pong double-buffered over s.
// ---------------------------------------------------------------------------
__global__ void __launch_bounds__(256, 1)
k_sp2p_softmax(const unsigned short* __restrict__ A,   // L2PT  [96,1024,1024]
               const unsigned short* __restrict__ Bt,  // L2SPT [96,1024,1024]
               unsigned short* __restrict__ out) {     // SP2P  [96,1024,1024]
  __shared__ float tile[16][1024];
  const int lane = threadIdx.x & 31;
  const int wave = threadIdx.x >> 5;
  const int pb = blockIdx.x;   // 0..31, 32 rows each
  const int bh = blockIdx.y;

  const unsigned short* AbLo = A + ((size_t)bh * 1024 + pb * 32) * 1024;
  const unsigned short* AbHi = AbLo + (size_t)16 * 1024;
  const unsigned short* Bw = Bt + ((size_t)bh * 1024 + wave * 128) * 1024;

  const v8f zero = {0.f, 0.f, 0.f, 0.f, 0.f, 0.f, 0.f, 0.f};
  v8f accLo[8], accHi[8];
#pragma unroll
  for (int c = 0; c < 8; ++c) { accLo[c] = zero; accHi[c] = zero; }

  // prologue: buffer 0 <- ks = 0
  bf16x16 aLo0 = load_frag_a(AbLo, 1024, lane);
  bf16x16 aHi0 = load_frag_a(AbHi, 1024, lane);
  bf16x16 b0[8], b1[8];
  bf16x16 aLo1, aHi1;
#pragma unroll
  for (int c = 0; c < 8; ++c) b0[c] = load_frag_b(Bw + (size_t)c * 16 * 1024, 1024, lane);

  for (int k0 = 0; k0 < 1024; k0 += 64) {
    const int k1 = k0 + 32;
    // issue loads for k1 into buffer 1
    aLo1 = load_frag_a(AbLo + k1, 1024, lane);
    aHi1 = load_frag_a(AbHi + k1, 1024, lane);
#pragma unroll
    for (int c = 0; c < 8; ++c)
      b1[c] = load_frag_b(Bw + (size_t)c * 16 * 1024 + k1, 1024, lane);
    // compute with buffer 0 (k0): each B fragment feeds two WMMAs
#pragma unroll
    for (int c = 0; c < 8; ++c) {
      accLo[c] = wmma_bf16(aLo0, b0[c], accLo[c]);
      accHi[c] = wmma_bf16(aHi0, b0[c], accHi[c]);
    }
    // issue loads for k0+64 into buffer 0
    const int k2 = k0 + 64;
    if (k2 < 1024) {
      aLo0 = load_frag_a(AbLo + k2, 1024, lane);
      aHi0 = load_frag_a(AbHi + k2, 1024, lane);
#pragma unroll
      for (int c = 0; c < 8; ++c)
        b0[c] = load_frag_b(Bw + (size_t)c * 16 * 1024 + k2, 1024, lane);
    }
    // compute with buffer 1 (k1)
#pragma unroll
    for (int c = 0; c < 8; ++c) {
      accLo[c] = wmma_bf16(aLo1, b1[c], accLo[c]);
      accHi[c] = wmma_bf16(aHi1, b1[c], accHi[c]);
    }

    // pin the pipeline: loads(k1) | wmma(k0) | loads(k2) | wmma(k1)
    SCHED_GROUP(0x020, 20, 0);
    SCHED_GROUP(0x008, 16, 0);
    SCHED_GROUP(0x020, 20, 0);
    SCHED_GROUP(0x008, 16, 0);
  }

  const int n = lane & 15, half = lane >> 4;
  const int row = threadIdx.x >> 4;
  const int sub = threadIdx.x & 15;

  // ---- phase 0: rows pb*32 .. pb*32+15 (accLo) ----
#pragma unroll
  for (int c = 0; c < 8; ++c)
#pragma unroll
    for (int r = 0; r < 8; ++r)
      tile[half * 8 + r][wave * 128 + c * 16 + n] = accLo[c][r];
  __syncthreads();
  {
    float mx = -1e30f;
    for (int i = sub; i < 1024; i += 16) mx = fmaxf(mx, tile[row][i]);
    for (int m = 8; m >= 1; m >>= 1) mx = fmaxf(mx, __shfl_xor(mx, m, 32));
    float sum = 0.f;
    for (int i = sub; i < 1024; i += 16) {
      const float e = __expf(tile[row][i] - mx);
      tile[row][i] = e;
      sum += e;
    }
    for (int m = 8; m >= 1; m >>= 1) sum += __shfl_xor(sum, m, 32);
    const float inv = 1.0f / sum;
    unsigned short* orow = out + ((size_t)bh * 1024 + pb * 32 + row) * 1024;
    for (int i = sub; i < 1024; i += 16) orow[i] = f2bf(tile[row][i] * inv);
  }
  __syncthreads();

  // ---- phase 1: rows pb*32+16 .. pb*32+31 (accHi) ----
#pragma unroll
  for (int c = 0; c < 8; ++c)
#pragma unroll
    for (int r = 0; r < 8; ++r)
      tile[half * 8 + r][wave * 128 + c * 16 + n] = accHi[c][r];
  __syncthreads();
  {
    float mx = -1e30f;
    for (int i = sub; i < 1024; i += 16) mx = fmaxf(mx, tile[row][i]);
    for (int m = 8; m >= 1; m >>= 1) mx = fmaxf(mx, __shfl_xor(mx, m, 32));
    float sum = 0.f;
    for (int i = sub; i < 1024; i += 16) {
      const float e = __expf(tile[row][i] - mx);
      tile[row][i] = e;
      sum += e;
    }
    for (int m = 8; m >= 1; m >>= 1) sum += __shfl_xor(sum, m, 32);
    const float inv = 1.0f / sum;
    unsigned short* orow = out + ((size_t)bh * 1024 + pb * 32 + 16 + row) * 1024;
    for (int i = sub; i < 1024; i += 16) orow[i] = f2bf(tile[row][i] * inv);
  }
}

// ---------------------------------------------------------------------------
// K_t: VspT[bh][d][k] = Vsp[b][k][h*64+d]   (bf16 transpose, 6.3M elems)
// ---------------------------------------------------------------------------
__global__ void k_transpose_v(const unsigned short* __restrict__ Vsp,
                              unsigned short* __restrict__ Vt) {
  const size_t total = (size_t)96 * 64 * 1024;
  size_t i = (size_t)blockIdx.x * 256 + threadIdx.x;
  if (i >= total) return;
  const int k = (int)(i & 1023);
  const size_t t = i >> 10;
  const int d = (int)(t & 63);
  const int bh = (int)(t >> 6);
  const int b = bh / 12, h = bh % 12;
  Vt[i] = Vsp[((size_t)b * 1024 + k) * 768 + h * 64 + d];
}

// ---------------------------------------------------------------------------
// K4: OH[b][p][h*64+d] = sum_k SP2P[bh][p][k] * VspT[bh][d][k]
// grid = (8 p-blocks of 128, 96 bh), 256 threads. Ping-pong over k.
// ---------------------------------------------------------------------------
__global__ void __launch_bounds__(256)
k_attn_out(const unsigned short* __restrict__ S,   // [96,1024,1024]
           const unsigned short* __restrict__ Vt,  // [96,64,1024]
           unsigned short* __restrict__ OH) {      // [8,1024,768]
  const int lane = threadIdx.x & 31;
  const int wave = threadIdx.x >> 5;
  const int pb = blockIdx.x;
  const int bh = blockIdx.y;
  const int b = bh / 12, h = bh % 12;
  const int row0 = pb * 128 + wave * 16;

  const unsigned short* Sb = S + ((size_t)bh * 1024 + row0) * 1024;
  const unsigned short* Vb = Vt + (size_t)bh * 64 * 1024;

  const v8f zero = {0.f, 0.f, 0.f, 0.f, 0.f, 0.f, 0.f, 0.f};
  v8f acc[4];
#pragma unroll
  for (int c = 0; c < 4; ++c) acc[c] = zero;

  bf16x16 a0 = load_frag_a(Sb, 1024, lane);
  bf16x16 b0[4], b1[4];
  bf16x16 a1;
#pragma unroll
  for (int c = 0; c < 4; ++c) b0[c] = load_frag_b(Vb + (size_t)c * 16 * 1024, 1024, lane);

  for (int k0 = 0; k0 < 1024; k0 += 64) {
    const int k1 = k0 + 32;
    a1 = load_frag_a(Sb + k1, 1024, lane);
#pragma unroll
    for (int c = 0; c < 4; ++c)
      b1[c] = load_frag_b(Vb + (size_t)c * 16 * 1024 + k1, 1024, lane);
#pragma unroll
    for (int c = 0; c < 4; ++c) acc[c] = wmma_bf16(a0, b0[c], acc[c]);
    const int k2 = k0 + 64;
    if (k2 < 1024) {
      a0 = load_frag_a(Sb + k2, 1024, lane);
#pragma unroll
      for (int c = 0; c < 4; ++c)
        b0[c] = load_frag_b(Vb + (size_t)c * 16 * 1024 + k2, 1024, lane);
    }
#pragma unroll
    for (int c = 0; c < 4; ++c) acc[c] = wmma_bf16(a1, b1[c], acc[c]);

    // pin the pipeline: loads(k1) | wmma(k0) | loads(k2) | wmma(k1)
    SCHED_GROUP(0x020, 10, 0);
    SCHED_GROUP(0x008, 4, 0);
    SCHED_GROUP(0x020, 10, 0);
    SCHED_GROUP(0x008, 4, 0);
  }

  const int n = lane & 15, half = lane >> 4;
#pragma unroll
  for (int c = 0; c < 4; ++c)
#pragma unroll
    for (int r = 0; r < 8; ++r) {
      const size_t prow = (size_t)(row0 + half * 8 + r);
      OH[((size_t)b * 1024 + prow) * 768 + h * 64 + c * 16 + n] = f2bf(acc[c][r]);
    }
}

// ---------------------------------------------------------------------------
// Host launcher
// ---------------------------------------------------------------------------
extern "C" void kernel_launch(void* const* d_in, const int* in_sizes, int n_in,
                              void* d_out, int out_size, void* d_ws, size_t ws_size,
                              hipStream_t stream) {
  (void)in_sizes; (void)n_in; (void)out_size; (void)ws_size;
  const float* l    = (const float*)d_in[0];
  const float* sp   = (const float*)d_in[1];
  const float* p    = (const float*)d_in[2];
  const float* Wq   = (const float*)d_in[3];
  const float* bq   = (const float*)d_in[4];
  const float* Wksp = (const float*)d_in[5];
  const float* bksp = (const float*)d_in[6];
  const float* Wkp  = (const float*)d_in[7];
  const float* bkp  = (const float*)d_in[8];
  const float* Wvsp = (const float*)d_in[9];
  const float* bvsp = (const float*)d_in[10];
  // d_in[11], d_in[12] (Wvp, bvp) are unused by the reference
  const float* Wo   = (const float*)d_in[13];
  const float* bo   = (const float*)d_in[14];
  float* out = (float*)d_out;

  const size_t NTOK = 8 * 1024;  // B * 1024
  const size_t E = 768;
  const float scale = 0.125f;    // 64^-0.5

  char* ws = (char*)d_ws;
  size_t off = 0;
  auto take = [&](size_t bytes) -> void* {
    void* ptr = ws + off;
    off += (bytes + 255) & ~(size_t)255;
    return ptr;
  };
  unsigned short* lbf   = (unsigned short*)take(NTOK * E * 2);
  unsigned short* spbf  = (unsigned short*)take(NTOK * E * 2);
  unsigned short* pbf   = (unsigned short*)take(NTOK * E * 2);
  unsigned short* wqb   = (unsigned short*)take(E * E * 2);
  unsigned short* wkspb = (unsigned short*)take(E * E * 2);
  unsigned short* wkpb  = (unsigned short*)take(E * E * 2);
  unsigned short* wvspb = (unsigned short*)take(E * E * 2);
  unsigned short* wob   = (unsigned short*)take(E * E * 2);
  unsigned short* Q     = (unsigned short*)take(NTOK * E * 2);
  unsigned short* Ksp   = (unsigned short*)take(NTOK * E * 2);
  unsigned short* Kp    = (unsigned short*)take(NTOK * E * 2);
  unsigned short* Vsp   = (unsigned short*)take(NTOK * E * 2);
  unsigned short* VspT  = (unsigned short*)take((size_t)96 * 64 * 1024 * 2);
  unsigned short* L2SPT = (unsigned short*)take((size_t)96 * 1024 * 1024 * 2);
  unsigned short* L2PT  = (unsigned short*)take((size_t)96 * 1024 * 1024 * 2);
  unsigned short* SP2P  = (unsigned short*)take((size_t)96 * 1024 * 1024 * 2);
  unsigned short* OH    = (unsigned short*)take(NTOK * E * 2);

  const dim3 blk(256);
  const int nTokE = (int)(NTOK * E);
  const int nWE = (int)(E * E);

  // K0: bf16 conversions
  k_f32_to_bf16<<<dim3((nTokE + 255) / 256), blk, 0, stream>>>(l, lbf, nTokE);
  k_f32_to_bf16<<<dim3((nTokE + 255) / 256), blk, 0, stream>>>(sp, spbf, nTokE);
  k_f32_to_bf16<<<dim3((nTokE + 255) / 256), blk, 0, stream>>>(p, pbf, nTokE);
  k_f32_to_bf16<<<dim3((nWE + 255) / 256), blk, 0, stream>>>(Wq, wqb, nWE);
  k_f32_to_bf16<<<dim3((nWE + 255) / 256), blk, 0, stream>>>(Wksp, wkspb, nWE);
  k_f32_to_bf16<<<dim3((nWE + 255) / 256), blk, 0, stream>>>(Wkp, wkpb, nWE);
  k_f32_to_bf16<<<dim3((nWE + 255) / 256), blk, 0, stream>>>(Wvsp, wvspb, nWE);
  k_f32_to_bf16<<<dim3((nWE + 255) / 256), blk, 0, stream>>>(Wo, wob, nWE);

  // K1: projections  proj = (x @ W^T + b) * scale, bf16 out
  const dim3 gProj(512);  // 8192 rows / 16
  k_gemm768<false><<<gProj, blk, 0, stream>>>(lbf,  wqb,   bq,   scale, Q,   nullptr, 768);
  k_gemm768<false><<<gProj, blk, 0, stream>>>(spbf, wkspb, bksp, scale, Ksp, nullptr, 768);
  k_gemm768<false><<<gProj, blk, 0, stream>>>(pbf,  wkpb,  bkp,  scale, Kp,  nullptr, 768);
  k_gemm768<false><<<gProj, blk, 0, stream>>>(spbf, wvspb, bvsp, scale, Vsp, nullptr, 768);

  // K_t: transpose v_sp per head for the attn-output GEMM
  {
    const size_t tot = (size_t)96 * 64 * 1024;
    k_transpose_v<<<dim3((unsigned)((tot + 255) / 256)), blk, 0, stream>>>(Vsp, VspT);
  }

  // K2: transposed logits + softmax over S (the query axis)
  const dim3 gLog(64, 96);
  k_logits_softmax<<<gLog, blk, 0, stream>>>(Ksp, Q, L2SPT);
  k_logits_softmax<<<gLog, blk, 0, stream>>>(Kp,  Q, L2PT);

  // K3: dominant einsum + softmax over K (32 rows per WG)
  k_sp2p_softmax<<<dim3(32, 96), blk, 0, stream>>>(L2PT, L2SPT, SP2P);

  // K4: attention output per head -> concatenated heads
  k_attn_out<<<dim3(8, 96), blk, 0, stream>>>(SP2P, VspT, OH);

  // K5: final projection, f32 out with bias
  k_gemm768<true><<<gProj, blk, 0, stream>>>(OH, wob, bo, 1.0f, nullptr, out, 768);
}
